// M2A_84275848282236
// MI455X (gfx1250) — matrix operand order repaired and checked by reading, not compiled, not run on hardware
//
#include <hip/hip_runtime.h>

typedef __attribute__((ext_vector_type(16))) __bf16 v16bf;
typedef __attribute__((ext_vector_type(8)))  float  v8f;
typedef __attribute__((ext_vector_type(4)))  unsigned int u32x4;

union FragU { v16bf v; u32x4 u[2]; };

constexpr int  kD   = 128;
constexpr float kEps = 1e-5f;

__device__ __forceinline__ unsigned short f2bfu(float x) {
  union { float f; unsigned u; } c; c.f = x;
  unsigned r = c.u + 0x7FFFu + ((c.u >> 16) & 1u);
  return (unsigned short)(r >> 16);
}

// ---------------------------------------------------------------------------
// Edge kernel: per-edge pipeline with 3 bf16 WMMA GEMMs + in-register GroupNorm
// ---------------------------------------------------------------------------
__global__ __launch_bounds__(256) void att_edge_kernel(
    const float* __restrict__ actor_ctrs, const float* __restrict__ node_ctrs,
    const int* __restrict__ hi, const int* __restrict__ wi, int E,
    const float* __restrict__ Wd1, const float* __restrict__ bd1,
    const float* __restrict__ Wd2,
    const float* __restrict__ gnd2w, const float* __restrict__ gnd2b,
    const float* __restrict__ Wc1,
    const float* __restrict__ gnc1w, const float* __restrict__ gnc1b,
    const float* __restrict__ Wc2,
    const float* __restrict__ qc, const float* __restrict__ nc,
    float* __restrict__ accum)
{
  extern __shared__ __align__(16) char smem[];
  unsigned short* wd2  = (unsigned short*)smem;      // 128x128 bf16
  unsigned short* wc1a = wd2 + kD * kD;              // 128x128 bf16 (cols 0..127 of Wc1)
  unsigned short* wc2s = wc1a + kD * kD;             // 128x128 bf16
  float* gnp  = (float*)(wc2s + kD * kD);            // gnd2w | gnd2b | gnc1w | gnc1b
  float* wd1s = gnp + 4 * kD;                        // Wd1 col0 | Wd1 col1 | bd1
  unsigned short* tiles = (unsigned short*)(wd1s + 3 * kD); // 8 waves * 16x128 bf16

  const int tid = threadIdx.x;
  for (int idx = tid; idx < kD * kD; idx += 256) {
    int n = idx >> 7, k = idx & 127;
    wd2[idx]  = f2bfu(Wd2[idx]);
    wc1a[idx] = f2bfu(Wc1[n * 384 + k]);
    wc2s[idx] = f2bfu(Wc2[idx]);
  }
  if (tid < kD) {
    gnp[tid]          = gnd2w[tid];
    gnp[kD + tid]     = gnd2b[tid];
    gnp[2 * kD + tid] = gnc1w[tid];
    gnp[3 * kD + tid] = gnc1b[tid];
    wd1s[tid]          = Wd1[2 * tid + 0];
    wd1s[kD + tid]     = Wd1[2 * tid + 1];
    wd1s[2 * kD + tid] = bd1[tid];
  }
  __syncthreads();

  const int wave = tid >> 5;   // 8 waves of 32 (wave32)
  const int lane = tid & 31;
  const int m16  = lane & 15;  // edge row within tile / N within ntile
  const int g    = lane >> 4;  // half-wave group
  unsigned short* tile = tiles + wave * (16 * kD);

  const int numTiles = (E + 15) >> 4;

  v8f acc[8];

  auto run_gemm = [&](const unsigned short* W) {
    v8f z = {0.f, 0.f, 0.f, 0.f, 0.f, 0.f, 0.f, 0.f};
    #pragma unroll
    for (int nt = 0; nt < 8; ++nt) acc[nt] = z;
    #pragma unroll
    for (int kc = 0; kc < kD; kc += 32) {
      FragU a;  // A 16x32 bf16 layout: lane m16, 16 contiguous K starting kc+16g
      a.u[0] = *(const u32x4*)(tile + m16 * kD + kc + 16 * g);
      a.u[1] = *(const u32x4*)(tile + m16 * kD + kc + 16 * g + 8);
      #pragma unroll
      for (int nt = 0; nt < 8; ++nt) {
        FragU b;  // B 32x16 bf16 layout: lane n=16nt+m16, 16 contiguous K starting kc+16g
        const unsigned short* wp = W + (nt * 16 + m16) * kD + kc + 16 * g;
        b.u[0] = *(const u32x4*)(wp);
        b.u[1] = *(const u32x4*)(wp + 8);
        acc[nt] = __builtin_amdgcn_wmma_f32_16x16x32_bf16(
            false, a.v, false, b.v, (short)0, acc[nt], false, false);
      }
    }
  };

  auto gn_relu_store = [&](const float* gw, const float* gb) {
    float wv[8], bv[8];
    #pragma unroll
    for (int nt = 0; nt < 8; ++nt) { wv[nt] = gw[nt * 16 + m16]; bv[nt] = gb[nt * 16 + m16]; }
    #pragma unroll
    for (int r = 0; r < 8; ++r) {
      float s = 0.f, q = 0.f;
      #pragma unroll
      for (int nt = 0; nt < 8; ++nt) { float v = acc[nt][r]; s += v; q += v * v; }
      #pragma unroll
      for (int off = 1; off <= 8; off <<= 1) {  // reduce within 16-lane half-group
        s += __shfl_xor(s, off, 32);
        q += __shfl_xor(q, off, 32);
      }
      float mean = s * (1.f / 128.f);
      float var  = q * (1.f / 128.f) - mean * mean;
      float rs   = rsqrtf(var + kEps);
      int row = r + 8 * g;  // C layout: lanes 0-15 rows 0..7, lanes 16-31 rows 8..15
      #pragma unroll
      for (int nt = 0; nt < 8; ++nt) {
        float v = fmaxf((acc[nt][r] - mean) * rs * wv[nt] + bv[nt], 0.f);
        tile[row * kD + nt * 16 + m16] = f2bfu(v);
      }
    }
  };

  for (int base = blockIdx.x * 8; base < numTiles; base += gridDim.x * 8) {
    const int tIdx = base + wave;
    const bool tvalid = (tIdx < numTiles);
    const int tileBase = (tvalid ? tIdx : (numTiles - 1)) * 16;

    // stage 0: d = relu(ctr_d @ Wd1.T + bd1) -> tile (each lane: one edge, 64 feats)
    {
      int e = min(tileBase + m16, E - 1);
      int h = hi[e], w = wi[e];
      float c0 = actor_ctrs[2 * h]     - node_ctrs[2 * w];
      float c1 = actor_ctrs[2 * h + 1] - node_ctrs[2 * w + 1];
      int f0 = g * 64;
      #pragma unroll
      for (int j = 0; j < 64; j += 2) {
        int f = f0 + j;
        float va = fmaxf(c0 * wd1s[f]     + c1 * wd1s[kD + f]     + wd1s[2 * kD + f],     0.f);
        float vb = fmaxf(c0 * wd1s[f + 1] + c1 * wd1s[kD + f + 1] + wd1s[2 * kD + f + 1], 0.f);
        unsigned pk = (unsigned)f2bfu(va) | ((unsigned)f2bfu(vb) << 16);
        *(unsigned*)(tile + m16 * kD + f) = pk;
      }
    }
    __syncthreads();

    // stage 1: d2 = relu(gn(d @ Wd2.T))
    run_gemm(wd2);
    __syncthreads();
    gn_relu_store(gnp, gnp + kD);
    __syncthreads();

    // stage 2: cf = relu(gn(d2 @ Wc1a.T + qc[hi] + nc[wi]))
    run_gemm(wc1a);
    #pragma unroll
    for (int r = 0; r < 8; ++r) {
      int e = min(tileBase + r + 8 * g, E - 1);
      int h = hi[e], w = wi[e];
      const float* qr = qc + (size_t)h * kD;
      const float* nr = nc + (size_t)w * kD;
      #pragma unroll
      for (int nt = 0; nt < 8; ++nt) {
        int col = nt * 16 + m16;
        acc[nt][r] += qr[col] + nr[col];
      }
    }
    __syncthreads();
    gn_relu_store(gnp + 2 * kD, gnp + 3 * kD);
    __syncthreads();

    // stage 3: out = cf @ Wc2.T, scatter-add onto actors
    run_gemm(wc2s);
    #pragma unroll
    for (int r = 0; r < 8; ++r) {
      int e = tileBase + r + 8 * g;
      if (tvalid && e < E) {
        int h = hi[e];
        float* dst = accum + (size_t)h * kD + m16;
        #pragma unroll
        for (int nt = 0; nt < 8; ++nt)
          __hip_atomic_fetch_add(dst + nt * 16, acc[nt][r],
                                 __ATOMIC_RELAXED, __HIP_MEMORY_SCOPE_AGENT);
      }
    }
    __syncthreads();
  }
}

// ---------------------------------------------------------------------------
// Small M/N-level kernels (fp32 VALU; negligible FLOPs)
// ---------------------------------------------------------------------------
__device__ __forceinline__ void block_gn(float v, float* red, float& mean, float& rstd) {
  float s = v, q = v * v;
  #pragma unroll
  for (int off = 1; off < 32; off <<= 1) { s += __shfl_xor(s, off, 32); q += __shfl_xor(q, off, 32); }
  int wid = threadIdx.x >> 5;
  if ((threadIdx.x & 31) == 0) { red[wid] = s; red[4 + wid] = q; }
  __syncthreads();
  float S = red[0] + red[1] + red[2] + red[3];
  float Q = red[4] + red[5] + red[6] + red[7];
  mean = S * (1.f / 128.f);
  rstd = rsqrtf(Q * (1.f / 128.f) - mean * mean + kEps);
  __syncthreads();
}

__global__ __launch_bounds__(128) void node_pre_kernel(
    const float* __restrict__ nodes, const float* __restrict__ Wc1, float* __restrict__ nc)
{
  __shared__ float row[kD];
  int n = blockIdx.x, t = threadIdx.x;
  row[t] = nodes[(size_t)n * kD + t];
  __syncthreads();
  const float* w = Wc1 + (size_t)t * 384 + 256;
  float s = 0.f;
  #pragma unroll 8
  for (int k = 0; k < kD; ++k) s += row[k] * w[k];
  nc[(size_t)n * kD + t] = s;
}

__global__ __launch_bounds__(128) void actor_pre_kernel(
    const float* __restrict__ agts,
    const float* __restrict__ Wq, const float* __restrict__ gnqw, const float* __restrict__ gnqb,
    const float* __restrict__ Wc1, const float* __restrict__ Wa,
    float* __restrict__ qc, float* __restrict__ aW)
{
  __shared__ float row[kD];
  __shared__ float qrow[kD];
  __shared__ float red[8];
  int m = blockIdx.x, t = threadIdx.x;
  row[t] = agts[(size_t)m * kD + t];
  __syncthreads();
  const float* wq = Wq + (size_t)t * kD;
  float s = 0.f;
  #pragma unroll 8
  for (int k = 0; k < kD; ++k) s += row[k] * wq[k];
  float mean, rstd;
  block_gn(s, red, mean, rstd);
  qrow[t] = fmaxf((s - mean) * rstd * gnqw[t] + gnqb[t], 0.f);  // q = relu(gn(agts@Wq.T))
  __syncthreads();
  const float* wc = Wc1 + (size_t)t * 384 + 128;                // middle block of Wc1
  float s2 = 0.f;
  #pragma unroll 8
  for (int k = 0; k < kD; ++k) s2 += qrow[k] * wc[k];
  qc[(size_t)m * kD + t] = s2;
  const float* wa = Wa + (size_t)t * kD;
  float s3 = 0.f;
  #pragma unroll 8
  for (int k = 0; k < kD; ++k) s3 += row[k] * wa[k];
  aW[(size_t)m * kD + t] = s3;
}

__global__ __launch_bounds__(128) void finalize_kernel(
    const float* __restrict__ aW, const float* __restrict__ accum,
    const float* __restrict__ res,
    const float* __restrict__ gnnw, const float* __restrict__ gnnb,
    const float* __restrict__ Wl,
    const float* __restrict__ gnlw, const float* __restrict__ gnlb,
    float* __restrict__ out)
{
  __shared__ float arow[kD];
  __shared__ float red[8];
  int m = blockIdx.x, t = threadIdx.x;
  float a = aW[(size_t)m * kD + t] + accum[(size_t)m * kD + t];
  float mean, rstd;
  block_gn(a, red, mean, rstd);
  a = fmaxf((a - mean) * rstd * gnnw[t] + gnnb[t], 0.f);
  arow[t] = a;
  __syncthreads();
  const float* wl = Wl + (size_t)t * kD;
  float s = 0.f;
  #pragma unroll 8
  for (int k = 0; k < kD; ++k) s += arow[k] * wl[k];
  block_gn(s, red, mean, rstd);
  float l = (s - mean) * rstd * gnlw[t] + gnlb[t];
  out[(size_t)m * kD + t] = fmaxf(l + res[(size_t)m * kD + t], 0.f);
}

__global__ void zero_kernel(float* __restrict__ p, int n) {
  int i = blockIdx.x * 256 + threadIdx.x;
  if (i < n) p[i] = 0.f;
}

// ---------------------------------------------------------------------------
extern "C" void kernel_launch(void* const* d_in, const int* in_sizes, int n_in,
                              void* d_out, int out_size, void* d_ws, size_t ws_size,
                              hipStream_t stream) {
  (void)in_sizes; (void)n_in; (void)out_size; (void)ws_size;
  const int M = 4000, N = 20000, E = 150000;

  const float* actors     = (const float*)d_in[0];
  const float* nodes      = (const float*)d_in[1];
  const float* actor_ctrs = (const float*)d_in[2];
  const float* node_ctrs  = (const float*)d_in[3];
  const int*   hi         = (const int*)d_in[4];
  const int*   wi         = (const int*)d_in[5];
  const float* Wd1   = (const float*)d_in[6];
  const float* bd1   = (const float*)d_in[7];
  const float* Wd2   = (const float*)d_in[8];
  const float* gnd2w = (const float*)d_in[9];
  const float* gnd2b = (const float*)d_in[10];
  const float* Wq    = (const float*)d_in[11];
  const float* gnqw  = (const float*)d_in[12];
  const float* gnqb  = (const float*)d_in[13];
  const float* Wc1   = (const float*)d_in[14];
  const float* gnc1w = (const float*)d_in[15];
  const float* gnc1b = (const float*)d_in[16];
  const float* Wc2   = (const float*)d_in[17];
  const float* Wa    = (const float*)d_in[18];
  const float* gnnw  = (const float*)d_in[19];
  const float* gnnb  = (const float*)d_in[20];
  const float* Wl    = (const float*)d_in[21];
  const float* gnlw  = (const float*)d_in[22];
  const float* gnlb  = (const float*)d_in[23];

  float* agts_buf = (float*)d_ws;                 // M*128
  float* qc       = agts_buf + (size_t)M * kD;    // M*128
  float* aWp      = qc + (size_t)M * kD;          // M*128
  float* accum    = aWp + (size_t)M * kD;         // M*128
  float* nc       = accum + (size_t)M * kD;       // N*128

  constexpr size_t SMEM_EDGE =
      (size_t)3 * kD * kD * 2 +        // 3 bf16 weight matrices
      (size_t)(4 * kD + 3 * kD) * 4 +  // gn params + Wd1/bd1
      (size_t)8 * 16 * kD * 2;         // 8 wave-private bf16 tiles

  for (int i = 0; i < 2; ++i) {
    const float* Wd1_i   = Wd1 + (size_t)i * kD * 2;
    const float* bd1_i   = bd1 + (size_t)i * kD;
    const float* Wd2_i   = Wd2 + (size_t)i * kD * kD;
    const float* gnd2w_i = gnd2w + (size_t)i * kD;
    const float* gnd2b_i = gnd2b + (size_t)i * kD;
    const float* Wq_i    = Wq + (size_t)i * kD * kD;
    const float* gnqw_i  = gnqw + (size_t)i * kD;
    const float* gnqb_i  = gnqb + (size_t)i * kD;
    const float* Wc1_i   = Wc1 + (size_t)i * kD * 384;
    const float* gnc1w_i = gnc1w + (size_t)i * kD;
    const float* gnc1b_i = gnc1b + (size_t)i * kD;
    const float* Wc2_i   = Wc2 + (size_t)i * kD * kD;
    const float* Wa_i    = Wa + (size_t)i * kD * kD;
    const float* gnnw_i  = gnnw + (size_t)i * kD;
    const float* gnnb_i  = gnnb + (size_t)i * kD;
    const float* Wl_i    = Wl + (size_t)i * kD * kD;
    const float* gnlw_i  = gnlw + (size_t)i * kD;
    const float* gnlb_i  = gnlb + (size_t)i * kD;

    const float* in_a = (i == 0) ? actors : agts_buf;
    float* out_a = (i == 1) ? (float*)d_out : agts_buf;

    node_pre_kernel<<<N, 128, 0, stream>>>(nodes, Wc1_i, nc);
    actor_pre_kernel<<<M, 128, 0, stream>>>(in_a, Wq_i, gnqw_i, gnqb_i, Wc1_i, Wa_i, qc, aWp);
    zero_kernel<<<(M * kD + 255) / 256, 256, 0, stream>>>(accum, M * kD);
    att_edge_kernel<<<512, 256, SMEM_EDGE, stream>>>(
        actor_ctrs, node_ctrs, hi, wi, E,
        Wd1_i, bd1_i, Wd2_i, gnd2w_i, gnd2b_i,
        Wc1_i, gnc1w_i, gnc1b_i, Wc2_i,
        qc, nc, accum);
    finalize_kernel<<<M, 128, 0, stream>>>(aWp, accum, in_a,
                                           gnnw_i, gnnb_i, Wl_i, gnlw_i, gnlb_i, out_a);
  }
}
